// PerceiverEncoder_68264210202902
// MI455X (gfx1250) — matrix operand order repaired and checked by reading, not compiled, not run on hardware
//
#include <hip/hip_runtime.h>
#include <cstdint>
#include <cstddef>

// ---------------------------------------------------------------------------
// Types for CDNA5 WMMA (wave32, 16x16x32 bf16 -> f32)
// ---------------------------------------------------------------------------
typedef __bf16 bf16_t;
typedef __bf16 v16bf __attribute__((ext_vector_type(16)));
typedef __bf16 v8bf  __attribute__((ext_vector_type(8)));
typedef float  v8f   __attribute__((ext_vector_type(8)));
typedef unsigned int u32x4 __attribute__((ext_vector_type(4)));
typedef int i32x4 __attribute__((ext_vector_type(4)));
typedef int i32x8 __attribute__((ext_vector_type(8)));

#if defined(__has_builtin)
#if __has_builtin(__builtin_amdgcn_tensor_load_to_lds)
#define HAVE_TDM 1
#endif
#endif
#ifndef HAVE_TDM
#define HAVE_TDM 0
#endif

#define USE_ASYNC_LDS 1

static __device__ __forceinline__ v8f v8f_zero() {
  v8f z;
#pragma unroll
  for (int i = 0; i < 8; ++i) z[i] = 0.f;
  return z;
}

static __device__ __forceinline__ v16bf cat8(v8bf lo, v8bf hi) {
  return __builtin_shufflevector(lo, hi, 0, 1, 2, 3, 4, 5, 6, 7,
                                 8, 9, 10, 11, 12, 13, 14, 15);
}

static __device__ __forceinline__ v8f wmma_bf16(v16bf a, v16bf b, v8f c) {
  // D = A(16x32 bf16) x B(32x16 bf16) + C(16x16 f32)
  return __builtin_amdgcn_wmma_f32_16x16x32_bf16(
      /*neg_a=*/false, a, /*neg_b=*/false, b,
      /*c_mod=*/(short)0, c, /*reuse_a=*/false, /*reuse_b=*/false);
}

// LDS byte offset of a generic pointer to __shared__ (flat addr low 32 bits).
static __device__ __forceinline__ unsigned lds_off(const void* p) {
  return (unsigned)(uintptr_t)p;
}

// ---------------------------------------------------------------------------
// Elementwise helpers
// ---------------------------------------------------------------------------
__global__ void k_convert(const float* __restrict__ X, bf16_t* __restrict__ Y, int n) {
  int i = blockIdx.x * blockDim.x + threadIdx.x;
  if (i < n) Y[i] = (bf16_t)X[i];
}

// W is [NOUT, KIN] row-major (out x in). Produce WT[KIN, NOUT] bf16 (K x N).
__global__ void k_transpose_convert(const float* __restrict__ W, bf16_t* __restrict__ WT,
                                    int NOUT, int KIN) {
  int idx = blockIdx.x * blockDim.x + threadIdx.x;
  if (idx >= NOUT * KIN) return;
  int n = idx / KIN, k = idx % KIN;
  WT[(size_t)k * NOUT + n] = (bf16_t)W[idx];
}

// x: [16,3,224,224] f32 -> Xp: [3136, 768] bf16 (row = b*196 + py*14 + px,
// col = c*256 + ph*16 + pw) which matches conv_w's [768][3*16*16] flattening.
__global__ void k_im2col(const float* __restrict__ x, bf16_t* __restrict__ Xp) {
  int idx = blockIdx.x * blockDim.x + threadIdx.x;
  if (idx >= 3136 * 768) return;
  int row = idx / 768, k = idx % 768;
  int b = row / 196, p = row % 196;
  int py = p / 14, px = p % 14;
  int c = k >> 8, ph = (k >> 4) & 15, pw = k & 15;
  float v = x[(((size_t)b * 3 + c) * 224 + (py * 16 + ph)) * 224 + (px * 16 + pw)];
  Xp[idx] = (bf16_t)v;
}

// ---------------------------------------------------------------------------
// WMMA GEMM: C[M,N] = A[M,K](bf16, row-major) * B[K,N](bf16, row-major)
// Block tile 64x128, 8 waves, each wave 16(M) x 64(N) = 4 accumulators.
// A tile staged into LDS by the Tensor Data Mover (fallback: cooperative).
// ---------------------------------------------------------------------------
template <bool BIAS, bool GELU_ACT, bool ROWSCALE, bool OUT_BF16>
__global__ void __launch_bounds__(256) gemm_wmma_kernel(
    const bf16_t* __restrict__ A, const bf16_t* __restrict__ B,
    const float* __restrict__ bias,
    const float* __restrict__ gate, int gate_ld, int gate_off,
    float* __restrict__ outF, bf16_t* __restrict__ outB,
    int M, int N, int K, int accFlag) {
  __shared__ bf16_t As[64 * 32];   // [64 rows][32 k]
  __shared__ bf16_t Bt[128 * 32];  // transposed: [128 n][32 k]

  const int tid  = threadIdx.x;
  const int lane = tid & 31;
  const int wave = tid >> 5;
  const int wr   = wave >> 1;  // 0..3 (row tile of 16)
  const int wc   = wave & 1;   // 0..1 (col half of 64)
  const int m0   = blockIdx.y * 64;
  const int n0   = blockIdx.x * 128;

  v8f acc[4];
#pragma unroll
  for (int t = 0; t < 4; ++t) acc[t] = v8f_zero();

  const int ar = tid >> 2, ac = (tid & 3) * 8;   // A: 64x32, 8 halves/thread
  const int bk = tid >> 3, bn = (tid & 7) * 16;  // B: 32x128, 16 halves/thread

  for (int k0 = 0; k0 < K; k0 += 32) {
    v16bf bv = *(const v16bf*)(B + (size_t)(k0 + bk) * N + n0 + bn);
#if !HAVE_TDM
    v8bf av = *(const v8bf*)(A + (size_t)(m0 + ar) * K + k0 + ac);
#endif
    __syncthreads();
#if HAVE_TDM
    if (wave == 0) {
      // Tensor DMA: 32(cols) x 64(rows) bf16 tile of A -> As (natural layout).
      uint64_t ga = (uint64_t)(uintptr_t)(A + (size_t)m0 * K + k0);
      unsigned td0 = (unsigned)K;       // tensor_dim0 (elements per row)
      unsigned td1 = 1u << 20;          // tensor_dim1 (rows, generous)
      u32x4 g0;
      g0[0] = 1u;                                       // count=1, user mode
      g0[1] = lds_off(&As[0]);                          // lds_addr (bytes)
      g0[2] = (unsigned)ga;                             // global_addr[31:0]
      g0[3] = (unsigned)((ga >> 32) & 0x01FFFFFFu) | (2u << 30);  // addr msbs | type=2
      i32x8 g1;
      g1[0] = (int)(1u << 16);                          // data_size=1 (2 bytes)
      g1[1] = (int)((td0 & 0xFFFFu) << 16);             // tensor_dim0 lo
      g1[2] = (int)(((td0 >> 16) & 0xFFFFu) | ((td1 & 0xFFFFu) << 16));
      g1[3] = (int)(((td1 >> 16) & 0xFFFFu) | (32u << 16));  // tile_dim0=32
      g1[4] = 64;                                       // tile_dim1=64, tile_dim2=0
      g1[5] = (int)td0;                                 // tensor_dim0_stride = K
      g1[6] = 0;
      g1[7] = 0;
      i32x4 gz;
      gz[0] = 0; gz[1] = 0; gz[2] = 0; gz[3] = 0;
      i32x8 gz8;
#pragma unroll
      for (int i = 0; i < 8; ++i) gz8[i] = 0;
      __builtin_amdgcn_tensor_load_to_lds(g0, g1, gz, gz, gz8, 0);
      __builtin_amdgcn_s_wait_tensorcnt((short)0);
    }
#else
    *(v8bf*)(As + ar * 32 + ac) = av;
#endif
#pragma unroll
    for (int j = 0; j < 16; ++j) Bt[(bn + j) * 32 + bk] = bv[j];
    __syncthreads();
    if (k0 + 32 < K) {
      __builtin_prefetch((const void*)(A + (size_t)(m0 + ar) * K + k0 + 32 + ac), 0, 1);
      __builtin_prefetch((const void*)(B + (size_t)(k0 + 32 + bk) * N + n0 + bn), 0, 1);
    }
    // A fragment (16x32): lane holds row (lane&15); two contiguous 8-half groups
    const int rowA = wr * 16 + (lane & 15);
    const int aoff = (lane >> 4) * 16;
    v8bf alo = *(const v8bf*)(As + rowA * 32 + aoff);
    v8bf ahi = *(const v8bf*)(As + rowA * 32 + 8 + aoff);
    v16bf af = cat8(alo, ahi);
#pragma unroll
    for (int t = 0; t < 4; ++t) {
      // B fragment (32x16): lane holds col (lane&15); 16 contiguous K halves.
      const int nb   = wc * 64 + t * 16 + (lane & 15);
      const int boff = (lane >> 4) * 16;
      v16bf bfv = *(const v16bf*)(Bt + nb * 32 + boff);
      acc[t] = wmma_bf16(af, bfv, acc[t]);
    }
  }

#pragma unroll
  for (int t = 0; t < 4; ++t) {
    const int col = n0 + wc * 64 + t * 16 + (lane & 15);
    float bvv = BIAS ? bias[col] : 0.f;
#pragma unroll
    for (int r = 0; r < 8; ++r) {
      const int row = m0 + wr * 16 + (lane >> 4) * 8 + r;
      float v = acc[t][r] + bvv;
      if (GELU_ACT) v = 0.5f * v * (1.f + erff(v * 0.70710678f));
      if (ROWSCALE) v *= gate[(size_t)row * gate_ld + gate_off];
      if (OUT_BF16) {
        outB[(size_t)row * N + col] = (bf16_t)v;
      } else {
        float prev = accFlag ? outF[(size_t)row * N + col] : 0.f;
        outF[(size_t)row * N + col] = prev + v;
      }
    }
  }
}

// ---------------------------------------------------------------------------
// Flash-style MHA: grid(NH, B), 256 threads (8 waves), 256 queries, head=96.
// K stored [SkvPad][96] (async global->LDS), V transposed [96][SkvPad].
// ---------------------------------------------------------------------------
__global__ void __launch_bounds__(256) attn_kernel(
    const bf16_t* __restrict__ Qg, long qBatchStride, int ldq,
    const bf16_t* __restrict__ KVg, long kvBatchStride, int ldkv, int kOff, int vOff,
    int Skv, int SkvPad,
    bf16_t* __restrict__ Og, float scale) {
  extern __shared__ char smem_raw[];
  bf16_t* Kl  = (bf16_t*)smem_raw;                 // [SkvPad][96]
  bf16_t* Vl  = Kl + (size_t)SkvPad * 96;          // [96][SkvPad]
  bf16_t* Pst = Vl + (size_t)96 * SkvPad;          // [8 waves][16*32]

  const int h    = blockIdx.x;
  const int b    = blockIdx.y;
  const int tid  = threadIdx.x;
  const int lane = tid & 31;
  const int wave = tid >> 5;

  const bf16_t* Kb = KVg + (size_t)b * kvBatchStride + kOff + h * 96;
  const bf16_t* Vb = KVg + (size_t)b * kvBatchStride + vOff + h * 96;

  for (int s = tid; s < SkvPad; s += 256) {
    if (s < Skv) {
#if USE_ASYNC_LDS
      const unsigned kl = lds_off(Kl + s * 96);
#pragma unroll
      for (int d = 0; d < 96; d += 8) {
        const uint64_t ga = (uint64_t)(uintptr_t)(Kb + (size_t)s * ldkv + d);
        const unsigned la = kl + (unsigned)(d * 2);
        asm volatile("global_load_async_to_lds_b128 %0, %1, off"
                     :: "v"(la), "v"(ga) : "memory");
      }
#else
#pragma unroll
      for (int d = 0; d < 96; d += 8) {
        v8bf kv = *(const v8bf*)(Kb + (size_t)s * ldkv + d);
        *(v8bf*)(Kl + s * 96 + d) = kv;
      }
#endif
#pragma unroll
      for (int d = 0; d < 96; d += 8) {
        v8bf vv = *(const v8bf*)(Vb + (size_t)s * ldkv + d);
#pragma unroll
        for (int j = 0; j < 8; ++j) Vl[(d + j) * SkvPad + s] = vv[j];
      }
    } else {
#pragma unroll
      for (int d = 0; d < 96; d += 8) {
        v8bf z;
#pragma unroll
        for (int j = 0; j < 8; ++j) z[j] = (bf16_t)0.f;
        *(v8bf*)(Kl + s * 96 + d) = z;
#pragma unroll
        for (int j = 0; j < 8; ++j) Vl[(d + j) * SkvPad + s] = (bf16_t)0.f;
      }
    }
  }
#if USE_ASYNC_LDS
  asm volatile("s_wait_asynccnt 0x0" ::: "memory");
#endif
  __syncthreads();

  const bf16_t* Qb = Qg + (size_t)b * qBatchStride + h * 96;
  bf16_t* Pw = Pst + wave * (16 * 32);
  const int nkt2 = SkvPad / 32;

  for (int qt = wave; qt < 16; qt += 8) {  // 256 queries / 16 per tile
    v16bf qf[3];
    const int qrow = qt * 16 + (lane & 15);
#pragma unroll
    for (int ks = 0; ks < 3; ++ks) {
      const bf16_t* qp = Qb + (size_t)qrow * ldq + ks * 32 + (lane >> 4) * 16;
      v8bf lo = *(const v8bf*)qp;
      v8bf hi = *(const v8bf*)(qp + 8);
      qf[ks] = cat8(lo, hi);
    }

    float mrun[8], lrun[8];
#pragma unroll
    for (int r = 0; r < 8; ++r) { mrun[r] = -1e30f; lrun[r] = 0.f; }
    v8f oacc[6];
#pragma unroll
    for (int j = 0; j < 6; ++j) oacc[j] = v8f_zero();

    for (int kt2 = 0; kt2 < nkt2; ++kt2) {
      v8f s0 = v8f_zero(), s1 = v8f_zero();
#pragma unroll
      for (int ks = 0; ks < 3; ++ks) {
#pragma unroll
        for (int t = 0; t < 2; ++t) {
          const int key = kt2 * 32 + t * 16 + (lane & 15);
          v16bf bfv = *(const v16bf*)(Kl + key * 96 + ks * 32 + (lane >> 4) * 16);
          if (t == 0) s0 = wmma_bf16(qf[ks], bfv, s0);
          else        s1 = wmma_bf16(qf[ks], bfv, s1);
        }
      }
      const int key0 = kt2 * 32 + (lane & 15);
      const int key1 = key0 + 16;
#pragma unroll
      for (int r = 0; r < 8; ++r) {
        float a0 = (key0 < Skv) ? s0[r] * scale : -1e30f;
        float a1 = (key1 < Skv) ? s1[r] * scale : -1e30f;
        float mx = fmaxf(a0, a1);
#pragma unroll
        for (int off = 1; off < 16; off <<= 1) mx = fmaxf(mx, __shfl_xor(mx, off, 16));
        float newm = fmaxf(mrun[r], mx);
        float corr = __expf(mrun[r] - newm);
        float p0 = __expf(a0 - newm);
        float p1 = __expf(a1 - newm);
        float ps = p0 + p1;
#pragma unroll
        for (int off = 1; off < 16; off <<= 1) ps += __shfl_xor(ps, off, 16);
        lrun[r] = lrun[r] * corr + ps;
        mrun[r] = newm;
#pragma unroll
        for (int j = 0; j < 6; ++j) oacc[j][r] *= corr;
        s0[r] = p0; s1[r] = p1;
      }
      // Stage probabilities (C layout) -> per-wave LDS -> A fragment layout.
#pragma unroll
      for (int r = 0; r < 8; ++r) {
        const int prow = (lane >> 4) * 8 + r;
        Pw[prow * 32 + (lane & 15)]      = (bf16_t)s0[r];
        Pw[prow * 32 + 16 + (lane & 15)] = (bf16_t)s1[r];
      }
      asm volatile("s_wait_dscnt 0" ::: "memory");  // wave-local LDS RAW
      v16bf pf;
      {
        const int prow = lane & 15;
        const bf16_t* pp = Pw + prow * 32 + (lane >> 4) * 16;
        v8bf lo = *(const v8bf*)pp;
        v8bf hi = *(const v8bf*)(pp + 8);
        pf = cat8(lo, hi);
      }
#pragma unroll
      for (int j = 0; j < 6; ++j) {
        const int dim = j * 16 + (lane & 15);
        v16bf bfv = *(const v16bf*)(Vl + (size_t)dim * SkvPad + kt2 * 32 + (lane >> 4) * 16);
        oacc[j] = wmma_bf16(pf, bfv, oacc[j]);
      }
    }
#pragma unroll
    for (int r = 0; r < 8; ++r) {
      float inv = 1.f / lrun[r];
      const int row = qt * 16 + (lane >> 4) * 8 + r;
#pragma unroll
      for (int j = 0; j < 6; ++j) {
        const int col = h * 96 + j * 16 + (lane & 15);
        Og[((size_t)b * 256 + row) * 768 + col] = (bf16_t)(oacc[j][r] * inv);
      }
    }
  }
}

// ---------------------------------------------------------------------------
// LayerNorm(x + res) with gamma/beta. One block per row, 256 threads, D=768.
// Writes both f32 and bf16 outputs. Safe when outF aliases res.
// ---------------------------------------------------------------------------
__global__ void __launch_bounds__(256) ln_kernel(
    const float* __restrict__ x, const float* __restrict__ res, int resBroadcast,
    const float* __restrict__ g, const float* __restrict__ beta,
    float* __restrict__ outF, bf16_t* __restrict__ outB) {
  __shared__ float red[256];
  const int row = blockIdx.x;
  const int tid = threadIdx.x;
  const int rrow = resBroadcast ? (row & 255) : row;
  float v[3];
#pragma unroll
  for (int i = 0; i < 3; ++i) {
    int d = tid + i * 256;
    v[i] = x[(size_t)row * 768 + d] + res[(size_t)rrow * 768 + d];
  }
  float s = v[0] + v[1] + v[2];
  red[tid] = s;
  __syncthreads();
  for (int off = 128; off > 0; off >>= 1) {
    if (tid < off) red[tid] += red[tid + off];
    __syncthreads();
  }
  float mean = red[0] * (1.f / 768.f);
  __syncthreads();
  float sq = 0.f;
#pragma unroll
  for (int i = 0; i < 3; ++i) { float d = v[i] - mean; sq += d * d; }
  red[tid] = sq;
  __syncthreads();
  for (int off = 128; off > 0; off >>= 1) {
    if (tid < off) red[tid] += red[tid + off];
    __syncthreads();
  }
  float inv = rsqrtf(red[0] * (1.f / 768.f) + 1e-5f);
#pragma unroll
  for (int i = 0; i < 3; ++i) {
    int d = tid + i * 256;
    float o = (v[i] - mean) * inv * g[d] + beta[d];
    outF[(size_t)row * 768 + d] = o;
    outB[(size_t)row * 768 + d] = (bf16_t)o;
  }
}

// ---------------------------------------------------------------------------
// MoE router: softmax -> top-2 of probabilities -> softmax over selected probs.
// ---------------------------------------------------------------------------
__global__ void router_kernel(const float* __restrict__ xf, const float* __restrict__ rw,
                              const float* __restrict__ rb, float* __restrict__ gate, int T) {
  int t = blockIdx.x * blockDim.x + threadIdx.x;
  if (t >= T) return;
  float lg[8];
#pragma unroll
  for (int e = 0; e < 8; ++e) lg[e] = rb[e];
  for (int d = 0; d < 768; ++d) {
    float xv = xf[(size_t)t * 768 + d];
#pragma unroll
    for (int e = 0; e < 8; ++e) lg[e] += xv * rw[d * 8 + e];
  }
  float mx = lg[0];
#pragma unroll
  for (int e = 1; e < 8; ++e) mx = fmaxf(mx, lg[e]);
  float p[8], sum = 0.f;
#pragma unroll
  for (int e = 0; e < 8; ++e) { p[e] = __expf(lg[e] - mx); sum += p[e]; }
#pragma unroll
  for (int e = 0; e < 8; ++e) p[e] /= sum;
  int i0 = 0;
#pragma unroll
  for (int e = 1; e < 8; ++e) if (p[e] > p[i0]) i0 = e;
  int i1 = (i0 == 0) ? 1 : 0;
#pragma unroll
  for (int e = 0; e < 8; ++e) if (e != i1 && e != i0 && p[e] > p[i1]) i1 = e;
  float e0 = __expf(p[i0]), e1 = __expf(p[i1]);
  float den = e0 + e1;
#pragma unroll
  for (int e = 0; e < 8; ++e) gate[(size_t)t * 8 + e] = 0.f;
  gate[(size_t)t * 8 + i0] = e0 / den;
  gate[(size_t)t * 8 + i1] = e1 / den;
}

// Final: out[b,d] = mean over 256 latents.
__global__ void mean_kernel(const float* __restrict__ latf, float* __restrict__ out) {
  int idx = blockIdx.x * blockDim.x + threadIdx.x;
  if (idx >= 16 * 768) return;
  int b = idx / 768, d = idx % 768;
  float s = 0.f;
  for (int t = 0; t < 256; ++t) s += latf[((size_t)b * 256 + t) * 768 + d];
  out[idx] = s * (1.f / 256.f);
}

// ---------------------------------------------------------------------------
// Host orchestration
// ---------------------------------------------------------------------------
extern "C" void kernel_launch(void* const* d_in, const int* in_sizes, int n_in,
                              void* d_out, int out_size, void* d_ws, size_t ws_size,
                              hipStream_t stream) {
  (void)in_sizes; (void)n_in; (void)out_size; (void)ws_size;
  const int Bsz = 16, D = 768, NL = 256, NH = 8, NP = 196, NB = 6, NE = 8, DFF = 1536;
  const int TOK = Bsz * NL;   // 4096
  const int MP  = Bsz * NP;   // 3136
  const int D3  = 3 * D;      // 2304
  const float scale = 0.10206207262f; // 1/sqrt(96)

  const float* x        = (const float*)d_in[0];
  const float* conv_w   = (const float*)d_in[1];
  const float* conv_b   = (const float*)d_in[2];
  const float* latents  = (const float*)d_in[3];
  const float* ca_in_w  = (const float*)d_in[4];
  const float* ca_in_b  = (const float*)d_in[5];
  const float* ca_out_w = (const float*)d_in[6];
  const float* ca_out_b = (const float*)d_in[7];
  const float* ca_g     = (const float*)d_in[8];
  const float* ca_beta  = (const float*)d_in[9];
  const float* blk_in_w  = (const float*)d_in[10];
  const float* blk_in_b  = (const float*)d_in[11];
  const float* blk_out_w = (const float*)d_in[12];
  const float* blk_out_b = (const float*)d_in[13];
  const float* blk_n1_g  = (const float*)d_in[14];
  const float* blk_n1_b  = (const float*)d_in[15];
  const float* blk_rw    = (const float*)d_in[16];
  const float* blk_rb    = (const float*)d_in[17];
  const float* blk_w1    = (const float*)d_in[18];
  const float* blk_b1    = (const float*)d_in[19];
  const float* blk_w2    = (const float*)d_in[20];
  const float* blk_b2    = (const float*)d_in[21];
  const float* blk_n2_g  = (const float*)d_in[22];
  const float* blk_n2_b  = (const float*)d_in[23];
  float* out = (float*)d_out;

  char* wp = (char*)d_ws;
  auto alloc = [&](size_t bytes) -> char* {
    char* p = wp;
    wp += (bytes + 255) & ~(size_t)255;
    return p;
  };

  bf16_t* Xp    = (bf16_t*)alloc((size_t)MP * D * 2);
  bf16_t* WcT   = (bf16_t*)alloc((size_t)D * D * 2);
  bf16_t* Pt    = (bf16_t*)alloc((size_t)MP * D * 2);
  bf16_t* WinT  = (bf16_t*)alloc((size_t)D * D3 * 2);
  bf16_t* WoT   = (bf16_t*)alloc((size_t)D * D * 2);
  bf16_t* Lb    = (bf16_t*)alloc((size_t)NL * D * 2);
  bf16_t* Qpack = (bf16_t*)alloc((size_t)NL * D3 * 2);
  bf16_t* Pack  = (bf16_t*)alloc((size_t)TOK * D3 * 2);   // also KV pack (MP<TOK)
  bf16_t* Ob    = (bf16_t*)alloc((size_t)TOK * D * 2);
  float*  AF    = (float*) alloc((size_t)TOK * D * 4);
  float*  latF  = (float*) alloc((size_t)TOK * D * 4);
  bf16_t* latB  = (bf16_t*)alloc((size_t)TOK * D * 2);
  float*  gate  = (float*) alloc((size_t)TOK * NE * 4);
  bf16_t* Hb    = (bf16_t*)alloc((size_t)TOK * DFF * 2);
  float*  moeF  = (float*) alloc((size_t)TOK * D * 4);
  bf16_t* W1b   = (bf16_t*)alloc((size_t)D * DFF * 2);
  bf16_t* W2b   = (bf16_t*)alloc((size_t)DFF * D * 2);

  auto cdiv = [](int a, int b) { return (a + b - 1) / b; };

  // 1) Patchify: im2col + GEMM(Xp @ conv_w^T + conv_b) -> Pt (bf16)
  k_im2col<<<cdiv(MP * D, 256), 256, 0, stream>>>(x, Xp);
  k_transpose_convert<<<cdiv(D * D, 256), 256, 0, stream>>>(conv_w, WcT, D, D);
  gemm_wmma_kernel<true, false, false, true><<<dim3(D / 128, MP / 64), 256, 0, stream>>>(
      Xp, WcT, conv_b, nullptr, 0, 0, nullptr, Pt, MP, D, D, 0);

  // 2) Cross-attention QKV packs
  k_transpose_convert<<<cdiv(D3 * D, 256), 256, 0, stream>>>(ca_in_w, WinT, D3, D);
  k_convert<<<cdiv(NL * D, 256), 256, 0, stream>>>(latents, Lb, NL * D);
  gemm_wmma_kernel<true, false, false, true><<<dim3(D3 / 128, NL / 64), 256, 0, stream>>>(
      Lb, WinT, ca_in_b, nullptr, 0, 0, nullptr, Qpack, NL, D3, D, 0);
  gemm_wmma_kernel<true, false, false, true><<<dim3(D3 / 128, MP / 64), 256, 0, stream>>>(
      Pt, WinT, ca_in_b, nullptr, 0, 0, nullptr, Pack, MP, D3, D, 0);

  // 3) Cross attention (Skv = 196, padded to 224)
  {
    int SkvPad = 224;
    size_t smem = ((size_t)2 * SkvPad * 96 + 8 * 16 * 32) * sizeof(bf16_t);
    attn_kernel<<<dim3(NH, Bsz), 256, smem, stream>>>(
        Qpack, 0L, D3, Pack, (long)NP * D3, D3, D, 2 * D, NP, SkvPad, Ob, scale);
  }
  k_transpose_convert<<<cdiv(D * D, 256), 256, 0, stream>>>(ca_out_w, WoT, D, D);
  gemm_wmma_kernel<true, false, false, false><<<dim3(D / 128, TOK / 64), 256, 0, stream>>>(
      Ob, WoT, ca_out_b, nullptr, 0, 0, AF, nullptr, TOK, D, D, 0);
  ln_kernel<<<TOK, 256, 0, stream>>>(AF, latents, 1, ca_g, ca_beta, latF, latB);

  // 4) Transformer blocks
  for (int i = 0; i < NB; ++i) {
    // Self attention
    k_transpose_convert<<<cdiv(D3 * D, 256), 256, 0, stream>>>(
        blk_in_w + (size_t)i * D3 * D, WinT, D3, D);
    gemm_wmma_kernel<true, false, false, true><<<dim3(D3 / 128, TOK / 64), 256, 0, stream>>>(
        latB, WinT, blk_in_b + (size_t)i * D3, nullptr, 0, 0, nullptr, Pack, TOK, D3, D, 0);
    {
      int SkvPad = 256;
      size_t smem = ((size_t)2 * SkvPad * 96 + 8 * 16 * 32) * sizeof(bf16_t);
      attn_kernel<<<dim3(NH, Bsz), 256, smem, stream>>>(
          Pack, (long)NL * D3, D3, Pack, (long)NL * D3, D3, D, 2 * D, NL, SkvPad, Ob, scale);
    }
    k_transpose_convert<<<cdiv(D * D, 256), 256, 0, stream>>>(
        blk_out_w + (size_t)i * D * D, WoT, D, D);
    gemm_wmma_kernel<true, false, false, false><<<dim3(D / 128, TOK / 64), 256, 0, stream>>>(
        Ob, WoT, blk_out_b + (size_t)i * D, nullptr, 0, 0, AF, nullptr, TOK, D, D, 0);
    ln_kernel<<<TOK, 256, 0, stream>>>(AF, latF, 0, blk_n1_g + (size_t)i * D,
                                       blk_n1_b + (size_t)i * D, latF, latB);

    // MoE (dense over experts, gate-weighted accumulation)
    router_kernel<<<cdiv(TOK, 256), 256, 0, stream>>>(
        latF, blk_rw + (size_t)i * D * NE, blk_rb + (size_t)i * NE, gate, TOK);
    for (int e = 0; e < NE; ++e) {
      const size_t eoff = (size_t)(i * NE + e);
      k_convert<<<cdiv(D * DFF, 256), 256, 0, stream>>>(
          blk_w1 + eoff * D * DFF, W1b, D * DFF);
      gemm_wmma_kernel<true, true, false, true><<<dim3(DFF / 128, TOK / 64), 256, 0, stream>>>(
          latB, W1b, blk_b1 + eoff * DFF, nullptr, 0, 0, nullptr, Hb, TOK, DFF, D, 0);
      k_convert<<<cdiv(DFF * D, 256), 256, 0, stream>>>(
          blk_w2 + eoff * DFF * D, W2b, DFF * D);
      gemm_wmma_kernel<true, false, true, false><<<dim3(D / 128, TOK / 64), 256, 0, stream>>>(
          Hb, W2b, blk_b2 + eoff * D, gate, NE, e, moeF, nullptr, TOK, D, DFF, (e > 0) ? 1 : 0);
    }
    ln_kernel<<<TOK, 256, 0, stream>>>(moeF, latF, 0, blk_n2_g + (size_t)i * D,
                                       blk_n2_b + (size_t)i * D, latF, latB);
  }

  // 5) Mean over latents
  mean_kernel<<<cdiv(16 * 768, 256), 256, 0, stream>>>(latF, out);
}
